// SelfAttention_56401510531516
// MI455X (gfx1250) — matrix operand order repaired and checked by reading, not compiled, not run on hardware
//
#include <hip/hip_runtime.h>
#include <hip/hip_bf16.h>

#define BATCH 16
#define SEQ   2048
#define NE    1024
#define HS    128

typedef __bf16 bf16;
typedef __bf16 v16bf __attribute__((ext_vector_type(16)));
typedef float  v8f   __attribute__((ext_vector_type(8)));

struct U32x8 { uint4 a, b; };

__device__ __forceinline__ v8f wmma_bf16(v16bf a, v16bf b, v8f c) {
  // D = A(16x32 bf16) * B(32x16 bf16) + C(16x16 f32)
  return __builtin_amdgcn_wmma_f32_16x16x32_bf16(false, a, false, b, (short)0, c,
                                                 false, false);
}

// A operand (16xK=32, bf16). ISA 7.12.2: lanes 0-15 -> M=lane, K {0..7,16..23};
// lanes 16-31 -> M=lane-16, K {8..15,24..31}. base points at (row0, k0).
__device__ __forceinline__ v16bf load_tileA(const bf16* base, int stride, int lane) {
  const bf16* p = base + (size_t)(lane & 15) * stride + ((lane >> 4) << 3);
  uint4 lo = *(const uint4*)p;          // k +0..7
  uint4 hi = *(const uint4*)(p + 16);   // k +16..23
  U32x8 u{lo, hi};
  return __builtin_bit_cast(v16bf, u);
}

// B operand (K=32 x 16) given B^T stored row-major (rows = n, K contiguous):
// lanes 0-15 -> N=lane, K {0..7,16..23}; lanes 16-31 -> N=lane-16, K {8..15,24..31}.
__device__ __forceinline__ v16bf load_tileB(const bf16* base, int stride, int lane) {
  const bf16* p = base + (size_t)(lane & 15) * stride + ((lane >> 4) << 3);
  uint4 lo = *(const uint4*)p;          // k +0..7
  uint4 hi = *(const uint4*)(p + 16);   // k +16..23
  U32x8 u{lo, hi};
  return __builtin_bit_cast(v16bf, u);
}

// A operand built from fp32 source with on-the-fly bf16 conversion.
__device__ __forceinline__ v16bf load_tileA_f32(const float* base, int stride, int lane) {
  const float* p = base + (size_t)(lane & 15) * stride + ((lane >> 4) << 3);
  float4 f0 = *(const float4*)(p);
  float4 f1 = *(const float4*)(p + 4);
  float4 f2 = *(const float4*)(p + 16);
  float4 f3 = *(const float4*)(p + 20);
  v16bf v;
  v[0]=(bf16)f0.x;  v[1]=(bf16)f0.y;  v[2]=(bf16)f0.z;  v[3]=(bf16)f0.w;
  v[4]=(bf16)f1.x;  v[5]=(bf16)f1.y;  v[6]=(bf16)f1.z;  v[7]=(bf16)f1.w;
  v[8]=(bf16)f2.x;  v[9]=(bf16)f2.y;  v[10]=(bf16)f2.z; v[11]=(bf16)f2.w;
  v[12]=(bf16)f3.x; v[13]=(bf16)f3.y; v[14]=(bf16)f3.z; v[15]=(bf16)f3.w;
  return v;
}

// ---------------------------------------------------------------------------
// Kernel 0: W[k][n] fp32 -> W^T[n][k] bf16 for Wk, Wq, Wv.
// ---------------------------------------------------------------------------
__global__ void wtrans_kernel(const float* __restrict__ Wk, const float* __restrict__ Wq,
                              const float* __restrict__ Wv, bf16* __restrict__ wts) {
  int idx = blockIdx.x * blockDim.x + threadIdx.x;  // 0 .. HS*NE-1
  int g = blockIdx.y;                               // 0=k 1=q 2=v
  const float* W = (g == 0) ? Wk : ((g == 1) ? Wq : Wv);
  bf16* WT = wts + (size_t)g * (HS * NE);
  int n  = idx >> 10;     // / NE
  int kk = idx & 1023;    // % NE
  WT[(size_t)n * NE + kk] = (bf16)W[(size_t)kk * HS + n];
}

// ---------------------------------------------------------------------------
// Kernel 1: QKV projection. One wave owns TWO 16-row x tiles (32 rows); each B
// tile feeds two WMMAs (forcing it live across both) and B-load traffic per
// WMMA halves. N-tiles ping-pong in two groups of 4 so every WMMA group has an
// independent load clause issued ahead of it. (No extra unrolling: that was
// observed to explode register copies.)
// ---------------------------------------------------------------------------
__global__ void __launch_bounds__(128) qkv_kernel(
    const float* __restrict__ x, const bf16* __restrict__ wts,
    const float* __restrict__ bk, const float* __restrict__ bq,
    const float* __restrict__ bv,
    bf16* __restrict__ kbuf, bf16* __restrict__ qbuf, bf16* __restrict__ vT) {
  int lane = threadIdx.x & 31;
  int wid  = threadIdx.x >> 5;
  int mp = blockIdx.x * 4 + wid;     // 32-row pair tile: 0..1023
  int g = blockIdx.y;
  int m0 = mp << 5;

  const bf16* WTg = wts + (size_t)g * (HS * NE);
  const float* xb0 = x + (size_t)m0 * NE;
  const float* xb1 = xb0 + (size_t)16 * NE;

  v8f acc[2][8];
#pragma unroll
  for (int mi = 0; mi < 2; ++mi)
#pragma unroll
    for (int i = 0; i < 8; ++i) acc[mi][i] = (v8f){};

  // Prologue: A tiles for chunk 0 and B group 0.
  v16bf a0 = load_tileA_f32(xb0, NE, lane);
  v16bf a1 = load_tileA_f32(xb1, NE, lane);
  v16bf b0[4];
#pragma unroll
  for (int nt = 0; nt < 4; ++nt)
    b0[nt] = load_tileB(WTg + (size_t)(nt * 16) * NE, NE, lane);

#pragma unroll 1
  for (int kc = 0; kc < NE / 32; ++kc) {
    // Load B group 1 for this chunk (overlaps group-0 WMMAs below).
    v16bf b1[4];
#pragma unroll
    for (int nt = 0; nt < 4; ++nt)
      b1[nt] = load_tileB(WTg + (size_t)((nt + 4) * 16) * NE + kc * 32, NE, lane);

    // WMMA group 0 (operands loaded one phase earlier).
#pragma unroll
    for (int nt = 0; nt < 4; ++nt) {
      acc[0][nt] = wmma_bf16(a0, b0[nt], acc[0][nt]);
      acc[1][nt] = wmma_bf16(a1, b0[nt], acc[1][nt]);
    }

    // Prefetch next chunk: A tiles (+cvt) and B group 0.
    if (kc < NE / 32 - 1) {
      v16bf a0n = load_tileA_f32(xb0 + (kc + 1) * 32, NE, lane);
      v16bf a1n = load_tileA_f32(xb1 + (kc + 1) * 32, NE, lane);
#pragma unroll
      for (int nt = 0; nt < 4; ++nt)
        b0[nt] = load_tileB(WTg + (size_t)(nt * 16) * NE + (kc + 1) * 32, NE, lane);
      // WMMA group 1.
#pragma unroll
      for (int nt = 0; nt < 4; ++nt) {
        acc[0][nt + 4] = wmma_bf16(a0, b1[nt], acc[0][nt + 4]);
        acc[1][nt + 4] = wmma_bf16(a1, b1[nt], acc[1][nt + 4]);
      }
      a0 = a0n;
      a1 = a1n;
    } else {
#pragma unroll
      for (int nt = 0; nt < 4; ++nt) {
        acc[0][nt + 4] = wmma_bf16(a0, b1[nt], acc[0][nt + 4]);
        acc[1][nt + 4] = wmma_bf16(a1, b1[nt], acc[1][nt + 4]);
      }
    }
  }

  const float* bias = (g == 0) ? bk : ((g == 1) ? bq : bv);
  int half = lane >> 4;
  int col  = lane & 15;
  if (g < 2) {
    bf16* dst = (g == 0) ? kbuf : qbuf;
#pragma unroll
    for (int mi = 0; mi < 2; ++mi)
#pragma unroll
      for (int nt = 0; nt < 8; ++nt) {
        float bcol = bias[nt * 16 + col];
#pragma unroll
        for (int r = 0; r < 8; ++r) {
          int row = m0 + mi * 16 + r + 8 * half;  // C layout rows
          dst[(size_t)row * HS + nt * 16 + col] = (bf16)(acc[mi][nt][r] + bcol);
        }
      }
  } else {
#pragma unroll
    for (int mi = 0; mi < 2; ++mi)
#pragma unroll
      for (int nt = 0; nt < 8; ++nt) {
        float bcol = bias[nt * 16 + col];
        int h = nt * 16 + col;
#pragma unroll
        for (int r = 0; r < 8; ++r) {
          int row = m0 + mi * 16 + r + 8 * half;
          int b_ = row >> 11;            // / SEQ
          int t_ = row & 2047;           // % SEQ
          vT[(size_t)b_ * (HS * SEQ) + (size_t)h * SEQ + t_] =
              (bf16)(acc[mi][nt][r] + bcol);
        }
      }
  }
}

// ---------------------------------------------------------------------------
// Kernel 2: flash attention. One wave per (batch, 16-query tile); 32-key steps
// up to the causal boundary. Q pinned in registers, O is 16x128 fp32.
// V operand loads are issued before the softmax VALU block to hide latency.
// Tile ids interleave short/long causal trip counts for load balance.
// ---------------------------------------------------------------------------
__global__ void __launch_bounds__(128) attn_kernel(
    const bf16* __restrict__ qbuf, const bf16* __restrict__ kbuf,
    const bf16* __restrict__ vT, float* __restrict__ out) {
  __shared__ __align__(16) bf16 ptile[4][16 * 32];   // per-wave P staging (1KB)
  int lane = threadIdx.x & 31;
  int wid  = threadIdx.x >> 5;
  int idx = blockIdx.x * 4 + wid;
  int b = idx >> 7;                        // 128 q-tiles per batch
  int j = idx & 127;
  int qt = (j & 1) ? (127 - (j >> 1)) : (j >> 1);   // balance causal work
  int q0 = qt << 4;
  int half = lane >> 4;
  int col  = lane & 15;

  const bf16* Qb = qbuf + (size_t)(b * SEQ + q0) * HS;
  v16bf Q[4];
#pragma unroll
  for (int c = 0; c < 4; ++c) Q[c] = load_tileA(Qb + c * 32, HS, lane);

  v8f o[8];
#pragma unroll
  for (int i = 0; i < 8; ++i) o[i] = (v8f){};
  float m[8], l[8];
#pragma unroll
  for (int r = 0; r < 8; ++r) { m[r] = -__builtin_inff(); l[r] = 0.0f; }

  const float sc = 0.08838834764831845f;  // 1/sqrt(HS)
  bf16* lds = ptile[wid];
  const bf16* Vbase = vT + (size_t)b * (HS * SEQ);

  int nkt = ((q0 + 15) >> 5) + 1;         // 32-key tiles intersecting causal band
#pragma unroll 1
  for (int kt = 0; kt < nkt; ++kt) {
    int k0 = kt << 5;
    const bf16* Kb = kbuf + (size_t)(b * SEQ + k0) * HS;

    // ---- S = Q * K^T : group the 8 operand loads, then 8 WMMAs ----
    v16bf kb0[4], kb1[4];
#pragma unroll
    for (int c = 0; c < 4; ++c) {
      kb0[c] = load_tileB(Kb + c * 32, HS, lane);
      kb1[c] = load_tileB(Kb + (size_t)16 * HS + c * 32, HS, lane);
    }
    v8f s0 = (v8f){}, s1 = (v8f){};
#pragma unroll
    for (int c = 0; c < 4; ++c) {
      s0 = wmma_bf16(Q[c], kb0[c], s0);
      s1 = wmma_bf16(Q[c], kb1[c], s1);
    }

    // ---- Issue V loads early: latency hidden behind the softmax VALU ----
    v16bf vb[8];
#pragma unroll
    for (int hc = 0; hc < 8; ++hc)
      vb[hc] = load_tileB(Vbase + (size_t)(hc * 16) * SEQ + k0, SEQ, lane);

    // ---- Online softmax (fp32) ----
    bool edge = (k0 + 31) > (q0 + 15);
#pragma unroll
    for (int r = 0; r < 8; ++r) {
      float a0 = s0[r] * sc;
      float a1 = s1[r] * sc;
      if (edge) {
        int rowg = q0 + r + 8 * half;
        a0 = (k0 + col)      > rowg ? -__builtin_inff() : a0;
        a1 = (k0 + 16 + col) > rowg ? -__builtin_inff() : a1;
      }
      // row max across the 16 lanes holding this row
      float t = fmaxf(a0, a1);
#pragma unroll
      for (int msk = 8; msk; msk >>= 1) t = fmaxf(t, __shfl_xor(t, msk, 32));
      float mn    = fmaxf(m[r], t);
      float alpha = __expf(m[r] - mn);
      float p0 = __expf(a0 - mn);
      float p1 = __expf(a1 - mn);
      float rs = p0 + p1;
#pragma unroll
      for (int msk = 8; msk; msk >>= 1) rs += __shfl_xor(rs, msk, 32);
      l[r] = l[r] * alpha + rs;
      m[r] = mn;
#pragma unroll
      for (int hc = 0; hc < 8; ++hc) o[hc][r] *= alpha;
      int rowm = r + 8 * half;
      lds[rowm * 32 + col]      = (bf16)p0;
      lds[rowm * 32 + 16 + col] = (bf16)p1;
    }
    // intra-wave C-layout -> A-layout relayout through LDS
    asm volatile("s_wait_dscnt 0" ::: "memory");
    v16bf P = load_tileA(lds, 32, lane);

    // ---- O += P * V ----
#pragma unroll
    for (int hc = 0; hc < 8; ++hc)
      o[hc] = wmma_bf16(P, vb[hc], o[hc]);
  }

#pragma unroll
  for (int r = 0; r < 8; ++r) {
    float inv = 1.0f / l[r];
    size_t row = (size_t)(b * SEQ + q0 + r + 8 * half);
#pragma unroll
    for (int hc = 0; hc < 8; ++hc)
      out[row * HS + hc * 16 + col] = o[hc][r] * inv;
  }
}

// ---------------------------------------------------------------------------
extern "C" void kernel_launch(void* const* d_in, const int* in_sizes, int n_in,
                              void* d_out, int out_size, void* d_ws, size_t ws_size,
                              hipStream_t stream) {
  const float* x  = (const float*)d_in[0];
  const float* Wk = (const float*)d_in[1];
  const float* bk = (const float*)d_in[2];
  const float* Wq = (const float*)d_in[3];
  const float* bq = (const float*)d_in[4];
  const float* Wv = (const float*)d_in[5];
  const float* bv = (const float*)d_in[6];
  float* out = (float*)d_out;

  // Workspace layout (bf16 elements), total ~26 MB:
  bf16* ws   = (bf16*)d_ws;
  bf16* wts  = ws;                                         // 3 * 128*1024
  bf16* kbuf = ws   + (size_t)3 * HS * NE;                 // B*T*HS
  bf16* qbuf = kbuf + (size_t)BATCH * SEQ * HS;            // B*T*HS
  bf16* vT   = qbuf + (size_t)BATCH * SEQ * HS;            // B*HS*T

  dim3 g0((HS * NE) / 256, 3);
  wtrans_kernel<<<g0, 256, 0, stream>>>(Wk, Wq, Wv, wts);

  dim3 g1((BATCH * SEQ / 32) / 4, 3);                      // 4 waves/block, 32 rows/wave
  qkv_kernel<<<g1, 128, 0, stream>>>(x, wts, bk, bq, bv, kbuf, qbuf, vT);

  attn_kernel<<<(BATCH * (SEQ / 16)) / 4, 128, 0, stream>>>(qbuf, kbuf, vT, out);
}
